// TorchMD_HEAD_56642028699884
// MI455X (gfx1250) — compile-verified, matrix operand order checked by reading, not run on hardware
//
#include <hip/hip_runtime.h>

#define NND 20000
#define NE  640000
#define HH  256
#define ETILES (NE / 16)

typedef __attribute__((ext_vector_type(16))) __bf16 v16bf;
typedef __attribute__((ext_vector_type(8)))  float  v8f;

// ---- WMMA fragment helpers -------------------------------------------------
// 16-bit A 16x32 layout: lane l: m = l&15, khalf = l>>4.
// element e (0..15): k = (e&7) + (e>>3)*16 + khalf*8
__device__ __forceinline__ v16bf load_a_f32(const float* row, int kbase, int khalf) {
  const float* p0 = row + kbase + khalf * 8;
  v16bf a;
#pragma unroll
  for (int i = 0; i < 8; ++i) {
    a[i]     = (__bf16)p0[i];
    a[i + 8] = (__bf16)p0[i + 16];
  }
  return a;
}

__device__ __forceinline__ v16bf load_a_bf16(const __bf16* row, int kbase, int khalf) {
  const __bf16* p0 = row + kbase + khalf * 8;
  v16bf a;
#pragma unroll
  for (int i = 0; i < 8; ++i) { a[i] = p0[i]; a[i + 8] = p0[i + 16]; }
  return a;
}

#define WMMA_BF16(A, B, C) \
  __builtin_amdgcn_wmma_f32_16x16x32_bf16(false, (A), false, (B), (short)0, (C), false, false)

__device__ __forceinline__ float silu_f(float x) { return x / (1.f + __expf(-x)); }

// ---- weight packing: fp32 W[K][NC] (up to 3 concatenated col-blocks) -> bf16
// packed B fragments: dst[((kt*NT + nt)*32 + lane)*16 + e] = W[k][n]
__global__ void pack_weights(const float* __restrict__ A, const float* __restrict__ B,
                             const float* __restrict__ C, int K, int NC, int split1,
                             int split2, __bf16* __restrict__ dst) {
  int NT = NC >> 4;
  int total = K * NC;
  for (int idx = blockIdx.x * blockDim.x + threadIdx.x; idx < total;
       idx += gridDim.x * blockDim.x) {
    int e = idx & 15;
    int lane = (idx >> 4) & 31;
    int rest = idx >> 9;
    int nt = rest % NT;
    int kt = rest / NT;
    int n = nt * 16 + (lane & 15);
    int khalf = lane >> 4;
    int k = kt * 32 + (e & 7) + ((e >> 3) << 4) + khalf * 8;
    float w;
    if (n < split1)      w = A[k * split1 + n];
    else if (n < split2) w = B[k * (split2 - split1) + (n - split1)];
    else                 w = C[k * (NC - split2) + (n - split2)];
    dst[idx] = (__bf16)w;
  }
}

__global__ void pack_bias(const float* __restrict__ b1, const float* __restrict__ b2,
                          const float* __restrict__ b3, int n1, int n2, int n3,
                          float* __restrict__ dst) {
  int i = blockIdx.x * blockDim.x + threadIdx.x;
  int tot = n1 + n2 + n3;
  if (i >= tot) return;
  float v;
  if (i < n1)            v = b1[i];
  else if (i < n1 + n2)  v = b2[i - n1];
  else                   v = b3[i - n1 - n2];
  dst[i] = v;
}

__global__ void zero_f32(float* __restrict__ p, int n) {
  for (int i = blockIdx.x * blockDim.x + threadIdx.x; i < n; i += gridDim.x * blockDim.x)
    p[i] = 0.f;
}

// ---- LayerNorm: one wave per node, 8 elems/lane, bf16 out ------------------
__global__ void __launch_bounds__(256) ln_kernel(const float* __restrict__ x,
                                                 const float* __restrict__ lnw,
                                                 const float* __restrict__ lnb,
                                                 __bf16* __restrict__ xn) {
  int wave = (blockIdx.x * blockDim.x + threadIdx.x) >> 5;
  int lane = threadIdx.x & 31;
  if (wave >= NND) return;
  const float* row = x + (size_t)wave * HH;
  float vals[8];
  float s = 0.f;
#pragma unroll
  for (int i = 0; i < 8; ++i) { vals[i] = row[lane * 8 + i]; s += vals[i]; }
#pragma unroll
  for (int off = 16; off; off >>= 1) s += __shfl_xor(s, off, 32);
  float mu = s * (1.f / HH);
  float var = 0.f;
#pragma unroll
  for (int i = 0; i < 8; ++i) { float d = vals[i] - mu; var += d * d; }
#pragma unroll
  for (int off = 16; off; off >>= 1) var += __shfl_xor(var, off, 32);
  float rstd = rsqrtf(var * (1.f / HH) + 1e-5f);
  __bf16* orow = xn + (size_t)wave * HH;
#pragma unroll
  for (int i = 0; i < 8; ++i) {
    int c = lane * 8 + i;
    orow[c] = (__bf16)((vals[i] - mu) * rstd * lnw[c] + lnb[c]);
  }
}

// ---- QKV GEMM: xn[N,256] @ packed [256,1280] + bias -> qkv[N,1280] ---------
__global__ void __launch_bounds__(256) gemm_qkv(const __bf16* __restrict__ xn,
                                                const __bf16* __restrict__ Wp,
                                                const float* __restrict__ bias,
                                                float* __restrict__ out) {
  const int NT = 80;
  int gwave = (blockIdx.x * blockDim.x + threadIdx.x) >> 5;
  int lane = threadIdx.x & 31;
  int mt = gwave / NT, nt = gwave % NT;
  if (mt >= NND / 16) return;
  int khalf = lane >> 4, l16 = lane & 15;
  float bval = bias[nt * 16 + l16];
  v8f acc;
#pragma unroll
  for (int r = 0; r < 8; ++r) acc[r] = bval;
  const __bf16* arow = xn + (size_t)(mt * 16 + l16) * HH;
#pragma unroll
  for (int kt = 0; kt < 8; ++kt) {
    v16bf a = load_a_bf16(arow, kt * 32, khalf);
    v16bf b = *(const v16bf*)(Wp + ((size_t)(kt * NT + nt) * 32 + lane) * 16);
    acc = WMMA_BF16(a, b, acc);
  }
  float* orow = out + (size_t)(mt * 16 + khalf * 8) * 1280 + nt * 16 + l16;
#pragma unroll
  for (int r = 0; r < 8; ++r) orow[(size_t)r * 1280] = acc[r];
}

// ---- vec projection: vec[N,3,256] @ Wvec[256,768]; fused vec_dot & vec3 ----
__global__ void __launch_bounds__(256) gemm_vec(const float* __restrict__ vec,
                                                const __bf16* __restrict__ Wp,
                                                float* __restrict__ vec_dot,
                                                float* __restrict__ vec3) {
  const int NT = 48;
  int gwave = (blockIdx.x * blockDim.x + threadIdx.x) >> 5;
  int lane = threadIdx.x & 31;
  int mt = gwave >> 4, ct = gwave & 15;
  if (mt >= NND / 16) return;
  int khalf = lane >> 4, l16 = lane & 15;
  v8f vdot;
#pragma unroll
  for (int r = 0; r < 8; ++r) vdot[r] = 0.f;
#pragma unroll
  for (int axis = 0; axis < 3; ++axis) {
    v8f a1, a2, a3;
#pragma unroll
    for (int r = 0; r < 8; ++r) { a1[r] = 0.f; a2[r] = 0.f; a3[r] = 0.f; }
    const float* arow = vec + ((size_t)(mt * 16 + l16) * 3 + axis) * HH;
#pragma unroll
    for (int kt = 0; kt < 8; ++kt) {
      v16bf a = load_a_f32(arow, kt * 32, khalf);
      v16bf b1 = *(const v16bf*)(Wp + ((size_t)(kt * NT + ct) * 32 + lane) * 16);
      v16bf b2 = *(const v16bf*)(Wp + ((size_t)(kt * NT + ct + 16) * 32 + lane) * 16);
      v16bf b3 = *(const v16bf*)(Wp + ((size_t)(kt * NT + ct + 32) * 32 + lane) * 16);
      a1 = WMMA_BF16(a, b1, a1);
      a2 = WMMA_BF16(a, b2, a2);
      a3 = WMMA_BF16(a, b3, a3);
    }
#pragma unroll
    for (int r = 0; r < 8; ++r) vdot[r] += a1[r] * a2[r];
    float* o = vec3 + ((size_t)(mt * 16 + khalf * 8) * 3 + axis) * HH + ct * 16 + l16;
#pragma unroll
    for (int r = 0; r < 8; ++r) o[(size_t)r * 3 * HH] = a3[r];
  }
  float* od = vec_dot + (size_t)(mt * 16 + khalf * 8) * HH + ct * 16 + l16;
#pragma unroll
  for (int r = 0; r < 8; ++r) od[(size_t)r * HH] = vdot[r];
}

// ---- fused edge kernel: dk/dv WMMA -> LDS, attention, message scatter ------
__global__ void __launch_bounds__(256) edge_kernel(
    const float* __restrict__ f_ij, const __bf16* __restrict__ Wdp,
    const float* __restrict__ bias_d, const int* __restrict__ edge_index,
    const float* __restrict__ r_ij, const float* __restrict__ d_ij,
    const float* __restrict__ qkv, const float* __restrict__ vec,
    float* __restrict__ x_agg, float* __restrict__ vec_agg) {
  extern __shared__ float smem[];
  float* stage = smem;              // [16][1024] silu(dk|dv), fp32
  float* lattn = smem + 16 * 1024;  // [16][8]
  int tid = threadIdx.x;
  int wave = tid >> 5, lane = tid & 31;
  int khalf = lane >> 4, l16 = lane & 15;

  for (int tile = blockIdx.x; tile < ETILES; tile += gridDim.x) {
    int ebase = tile * 16;
    // phase 1: [16,64] @ [64,1024] WMMA, silu, stage in LDS
    const float* arow = f_ij + (size_t)(ebase + l16) * 64;
    v16bf a0 = load_a_f32(arow, 0, khalf);
    v16bf a1 = load_a_f32(arow, 32, khalf);
#pragma unroll
    for (int t = 0; t < 8; ++t) {
      int ct = wave * 8 + t;
      float bval = bias_d[ct * 16 + l16];
      v8f acc;
#pragma unroll
      for (int r = 0; r < 8; ++r) acc[r] = bval;
      v16bf b0 = *(const v16bf*)(Wdp + ((size_t)(0 * 64 + ct) * 32 + lane) * 16);
      v16bf b1 = *(const v16bf*)(Wdp + ((size_t)(1 * 64 + ct) * 32 + lane) * 16);
      acc = WMMA_BF16(a0, b0, acc);
      acc = WMMA_BF16(a1, b1, acc);
#pragma unroll
      for (int r = 0; r < 8; ++r)
        stage[(r + khalf * 8) * 1024 + ct * 16 + l16] = silu_f(acc[r]);
    }
    __syncthreads();
    // phase 2: per-(edge,head) attention with cosine cutoff
    if (tid < 128) {
      int e = tid >> 3, h = tid & 7;
      int eg = ebase + e;
      int srci = edge_index[eg];
      int dsti = edge_index[NE + eg];
      const float* qr = qkv + (size_t)dsti * 1280 + h * 32;
      const float* kr = qkv + (size_t)srci * 1280 + 256 + h * 32;
      const float* dkr = &stage[e * 1024 + h * 32];
      float s = 0.f;
#pragma unroll
      for (int i = 0; i < 32; ++i) s += qr[i] * kr[i] * dkr[i];
      float rr = r_ij[eg];
      float cut = 0.5f * (__cosf(rr * (3.14159265358979323f / 5.0f)) + 1.0f);
      cut = (rr < 5.0f) ? cut : 0.f;
      lattn[e * 8 + h] = silu_f(s) * cut;
    }
    __syncthreads();
    // phase 3: messages + scatter-add (fp32 atomics)
    {
      int e = tid >> 4, c16 = tid & 15;
      int eg = ebase + e;
      int srci = edge_index[eg];
      int dsti = edge_index[NE + eg];
      float dj0 = d_ij[eg * 3 + 0], dj1 = d_ij[eg * 3 + 1], dj2 = d_ij[eg * 3 + 2];
      const float* vrow = qkv + (size_t)srci * 1280 + 512;   // v[src], [8,96]
      const float* vecrow = vec + (size_t)srci * 3 * HH;      // vec[src], [3,256]
      float* xo = x_agg + (size_t)dsti * HH;
      float* vo = vec_agg + (size_t)dsti * 3 * HH;
      const float* dvs = &stage[e * 1024 + 256];
#pragma unroll
      for (int g = 0; g < 16; ++g) {
        int col = c16 + g * 16;          // 0..255 (h*32+hd)
        int h = col >> 5, hd = col & 31;
        int j = h * 96 + hd;             // flat [8,96] index
        float at = lattn[e * 8 + h];
        float xm = vrow[j] * dvs[j] * at;
        atomicAdd(xo + col, xm);
        float v1 = vrow[j + 32] * dvs[j + 32];
        float v2 = vrow[j + 64] * dvs[j + 64];
        atomicAdd(vo + col,          vecrow[col] * v1 + v2 * dj0);
        atomicAdd(vo + HH + col,     vecrow[HH + col] * v1 + v2 * dj1);
        atomicAdd(vo + 2 * HH + col, vecrow[2 * HH + col] * v1 + v2 * dj2);
      }
    }
    __syncthreads();
  }
}

// ---- output GEMM: x_agg @ Wo + bo, fused dx/dvec epilogue ------------------
__global__ void __launch_bounds__(256) gemm_out(
    const float* __restrict__ x_agg, const __bf16* __restrict__ Wop,
    const float* __restrict__ bias_o, const float* __restrict__ vec_dot,
    const float* __restrict__ vec3, const float* __restrict__ vec_agg,
    float* __restrict__ out) {
  const int NT = 48;
  int gwave = (blockIdx.x * blockDim.x + threadIdx.x) >> 5;
  int lane = threadIdx.x & 31;
  int mt = gwave >> 4, ct = gwave & 15;
  if (mt >= NND / 16) return;
  int khalf = lane >> 4, l16 = lane & 15;
  v8f o1, o2, o3;
  float b1v = bias_o[ct * 16 + l16];
  float b2v = bias_o[256 + ct * 16 + l16];
  float b3v = bias_o[512 + ct * 16 + l16];
#pragma unroll
  for (int r = 0; r < 8; ++r) { o1[r] = b1v; o2[r] = b2v; o3[r] = b3v; }
  const float* arow = x_agg + (size_t)(mt * 16 + l16) * HH;
#pragma unroll
  for (int kt = 0; kt < 8; ++kt) {
    v16bf a = load_a_f32(arow, kt * 32, khalf);
    v16bf b1 = *(const v16bf*)(Wop + ((size_t)(kt * NT + ct) * 32 + lane) * 16);
    v16bf b2 = *(const v16bf*)(Wop + ((size_t)(kt * NT + ct + 16) * 32 + lane) * 16);
    v16bf b3 = *(const v16bf*)(Wop + ((size_t)(kt * NT + ct + 32) * 32 + lane) * 16);
    o1 = WMMA_BF16(a, b1, o1);
    o2 = WMMA_BF16(a, b2, o2);
    o3 = WMMA_BF16(a, b3, o3);
  }
  int m0 = mt * 16 + khalf * 8;
  int col = ct * 16 + l16;
  float* dx = out;
  float* dvec = out + (size_t)NND * HH;
#pragma unroll
  for (int r = 0; r < 8; ++r) {
    int m = m0 + r;
    float vd = vec_dot[(size_t)m * HH + col];
    dx[(size_t)m * HH + col] = vd * o2[r] + o3[r];
#pragma unroll
    for (int axis = 0; axis < 3; ++axis) {
      size_t vi = ((size_t)m * 3 + axis) * HH + col;
      dvec[vi] = vec3[vi] * o1[r] + vec_agg[vi];
    }
  }
}

// ---- launcher --------------------------------------------------------------
extern "C" void kernel_launch(void* const* d_in, const int* in_sizes, int n_in,
                              void* d_out, int out_size, void* d_ws, size_t ws_size,
                              hipStream_t stream) {
  const float* x    = (const float*)d_in[0];
  const float* vec  = (const float*)d_in[1];
  const int* eidx   = (const int*)d_in[2];
  const float* r_ij = (const float*)d_in[3];
  const float* f_ij = (const float*)d_in[4];
  const float* d_ij = (const float*)d_in[5];
  const float* ln_w = (const float*)d_in[6];
  const float* ln_b = (const float*)d_in[7];
  const float* Wq   = (const float*)d_in[8];
  const float* bq   = (const float*)d_in[9];
  const float* Wk   = (const float*)d_in[10];
  const float* bk   = (const float*)d_in[11];
  const float* Wv   = (const float*)d_in[12];
  const float* bv   = (const float*)d_in[13];
  const float* Wo   = (const float*)d_in[14];
  const float* bo   = (const float*)d_in[15];
  const float* Wvec = (const float*)d_in[16];
  const float* Wdk  = (const float*)d_in[17];
  const float* bdk  = (const float*)d_in[18];
  const float* Wdv  = (const float*)d_in[19];
  const float* bdv  = (const float*)d_in[20];
  float* out = (float*)d_out;

  char* ws = (char*)d_ws;
  size_t off = 0;
  auto take = [&](size_t bytes) {
    char* p = ws + off;
    off = (off + bytes + 255) & ~(size_t)255;
    return p;
  };
  __bf16* xn      = (__bf16*)take((size_t)NND * HH * 2);
  float* qkv      = (float*)take((size_t)NND * 1280 * 4);
  float* vec_dot  = (float*)take((size_t)NND * HH * 4);
  float* vec3     = (float*)take((size_t)NND * 3 * HH * 4);
  float* agg      = (float*)take((size_t)NND * 1024 * 4);  // x_agg | vec_agg
  float* x_agg    = agg;
  float* vec_agg  = agg + (size_t)NND * HH;
  __bf16* Wqkv_p  = (__bf16*)take((size_t)256 * 1280 * 2);
  __bf16* Wd_p    = (__bf16*)take((size_t)64 * 1024 * 2);
  __bf16* Wvec_p  = (__bf16*)take((size_t)256 * 768 * 2);
  __bf16* Wo_p    = (__bf16*)take((size_t)256 * 768 * 2);
  float* bias_qkv = (float*)take(1280 * 4);
  float* bias_d   = (float*)take(1024 * 4);
  float* bias_o   = (float*)take(768 * 4);

  // prep: weight/bias packing + accumulator zeroing
  pack_weights<<<1280, 256, 0, stream>>>(Wq, Wk, Wv, 256, 1280, 256, 512, Wqkv_p);
  pack_weights<<<256, 256, 0, stream>>>(Wdk, Wdv, Wdv, 64, 1024, 256, 1024, Wd_p);
  pack_weights<<<768, 256, 0, stream>>>(Wvec, Wvec, Wvec, 256, 768, 768, 768, Wvec_p);
  pack_weights<<<768, 256, 0, stream>>>(Wo, Wo, Wo, 256, 768, 768, 768, Wo_p);
  pack_bias<<<5, 256, 0, stream>>>(bq, bk, bv, 256, 256, 768, bias_qkv);
  pack_bias<<<4, 256, 0, stream>>>(bdk, bdv, bdv, 256, 768, 0, bias_d);
  pack_bias<<<3, 256, 0, stream>>>(bo, bo, bo, 768, 0, 0, bias_o);
  zero_f32<<<2048, 256, 0, stream>>>(agg, NND * 1024);

  // node pipeline
  ln_kernel<<<NND / 8, 256, 0, stream>>>(x, ln_w, ln_b, xn);
  gemm_qkv<<<(NND / 16) * 80 / 8, 256, 0, stream>>>(xn, Wqkv_p, bias_qkv, qkv);
  gemm_vec<<<(NND / 16) * 16 / 8, 256, 0, stream>>>(vec, Wvec_p, vec_dot, vec3);

  // fused edge pipeline (dk/dv WMMA in LDS + attention + scatter)
  edge_kernel<<<4096, 256, (16 * 1024 + 128) * sizeof(float), stream>>>(
      f_ij, Wd_p, bias_d, eidx, r_ij, d_ij, qkv, vec, x_agg, vec_agg);

  // output projection + epilogue
  gemm_out<<<(NND / 16) * 16 / 8, 256, 0, stream>>>(x_agg, Wo_p, bias_o, vec_dot,
                                                    vec3, vec_agg, out);
}
